// EMDLayer_33054068310367
// MI455X (gfx1250) — matrix operand order repaired and checked by reading, not compiled, not run on hardware
//
#include <hip/hip_runtime.h>

typedef __attribute__((ext_vector_type(16))) _Float16 v16h;
typedef __attribute__((ext_vector_type(8)))  _Float16 v8h;
typedef __attribute__((ext_vector_type(8)))  float    v8f;

#define WAY   5
#define SHOT  5
#define QRY   15
#define NQ    (WAY * QRY)   /* 75 */
#define M     64            /* nodes per image */
#define D     640           /* feature dim per node */
#define C     (D * M)
#define KHALF 320           /* K-split so the f16 tile fits in <64KB static LDS */
#define QPITCH 328          /* halfs per LDS row: 16B aligned, low bank conflict */
#define SPITCH 65           /* float pitch for sim/K: stride-65 => conflict-free */

// -------------------------------------------------------------------------
// Kernel 1: one block per (t,w).  proto = mean over shots; emit:
//   pnT   [t][w][n][d]  f16   (normalized proto, transposed = WMMA-B layout)
//   pnorm [t][w][n]     f32   (||proto|| per node, to rebuild raw dots)
//   pmean [t][w][d]     f32   (proto.mean over n, for w1)
// -------------------------------------------------------------------------
__global__ __launch_bounds__(256)
void prep_support(const float* __restrict__ sup, _Float16* __restrict__ pnT,
                  float* __restrict__ pnorm, float* __restrict__ pmean) {
  __shared__ float    pmeanL[D];
  __shared__ float    pnsqL[M];
  __shared__ float    rnormL[M];
  __shared__ __align__(16) _Float16 buf[M * 128];

  const int bid = blockIdx.x;            // t*WAY + w
  const int tid = threadIdx.x;
  const int n   = tid & 63;
  const int dg  = tid >> 6;
  const size_t base = (size_t)bid * SHOT * C;

  for (int i = tid; i < D; i += 256) pmeanL[i] = 0.f;
  if (tid < M) pnsqL[tid] = 0.f;
  __syncthreads();

  float nsq = 0.f;
  for (int d = dg; d < D; d += 4) {
    float p = 0.f;
#pragma unroll
    for (int s = 0; s < SHOT; ++s)
      p += sup[base + ((size_t)s * D + d) * M + n];
    p *= (1.f / SHOT);
    nsq += p * p;
    float ps = p;                          // reduce over the 64 n-lanes
#pragma unroll
    for (int k = 16; k >= 1; k >>= 1) ps += __shfl_xor(ps, k, 32);
    if ((tid & 31) == 0) atomicAdd(&pmeanL[d], ps);
  }
  atomicAdd(&pnsqL[n], nsq);
  __syncthreads();

  if (tid < M) {
    float nn = sqrtf(pnsqL[tid]);
    rnormL[tid] = 1.f / (nn + 1e-8f);
    pnorm[(size_t)bid * M + tid] = nn;
  }
  for (int i = tid; i < D; i += 256)
    pmean[(size_t)bid * D + i] = pmeanL[i] * (1.f / 64.f);
  __syncthreads();

  // pass 2: re-read (L2 hot), normalize, write transposed in coalesced chunks
  for (int ch = 0; ch < D; ch += 128) {
    for (int dl = dg; dl < 128; dl += 4) {
      int d = ch + dl;
      float p = 0.f;
#pragma unroll
      for (int s = 0; s < SHOT; ++s)
        p += sup[base + ((size_t)s * D + d) * M + n];
      p *= (1.f / SHOT);
      buf[n * 128 + dl] = (_Float16)(p * rnormL[n]);
    }
    __syncthreads();
    for (int idx = tid; idx < M * 128; idx += 256) {
      int rn = idx >> 7, j = idx & 127;
      pnT[((size_t)bid * M + rn) * D + ch + j] = buf[idx];
    }
    __syncthreads();
  }
}

// -------------------------------------------------------------------------
// Kernel 2: one block per (t,q), 256 threads (8 waves).
// Fused: norms/means/w1  ->  WMMA sim for all 5 ways  ->  Sinkhorn  ->  logits
// -------------------------------------------------------------------------
__global__ __launch_bounds__(256)
void emd_main(const float* __restrict__ qf, const _Float16* __restrict__ pnT,
              const float* __restrict__ pnorm, const float* __restrict__ pmean,
              float* __restrict__ out) {
  // 41984B region: f16 qn half-tile during GEMM; sim+K (2*16640B) afterwards
  __shared__ __align__(16) unsigned char smemRaw[M * QPITCH * 2];
  __shared__ float qmeanL[D];
  __shared__ float w1L[WAY * M];
  __shared__ float qnsqL[M];
  __shared__ float rnormQ[M];
  __shared__ float abuf[M], bbuf[M], uL[M], vL[M];
  __shared__ float partL[4 * M];
  __shared__ float redL[8];

  const int bid  = blockIdx.x;          // tt*NQ + qi
  const int tt   = bid / NQ;
  const int tid  = threadIdx.x;
  const int lane = tid & 31;
  const int wv   = tid >> 5;            // wave 0..7
  const int m    = tid & 63;
  const int dg   = tid >> 6;
  const size_t qbase = (size_t)bid * C;

  // ---- pass 1: node norms, query mean, w1 = <q_node, proto_mean> ----
  for (int i = tid; i < D; i += 256) qmeanL[i] = 0.f;
  if (tid < M) qnsqL[tid] = 0.f;
  for (int i = tid; i < WAY * M; i += 256) w1L[i] = 0.f;
  __syncthreads();

  const float* pmB = pmean + (size_t)tt * WAY * D;
  float nsq = 0.f;
  float w1a[WAY] = {0.f, 0.f, 0.f, 0.f, 0.f};
  for (int d = dg; d < D; d += 4) {
    float v = qf[qbase + (size_t)d * M + m];
    nsq += v * v;
#pragma unroll
    for (int w = 0; w < WAY; ++w) w1a[w] += v * pmB[w * D + d];
    float ps = v;
#pragma unroll
    for (int k = 16; k >= 1; k >>= 1) ps += __shfl_xor(ps, k, 32);
    if (lane == 0) atomicAdd(&qmeanL[d], ps);
  }
  atomicAdd(&qnsqL[m], nsq);
#pragma unroll
  for (int w = 0; w < WAY; ++w) atomicAdd(&w1L[w * M + m], w1a[w]);
  __syncthreads();

  if (tid < M) rnormQ[tid] = 1.f / (sqrtf(qnsqL[tid]) + 1e-8f);
  for (int i = tid; i < D; i += 256) qmeanL[i] *= (1.f / 64.f);
  for (int i = tid; i < WAY * M; i += 256)
    w1L[i] = fmaxf(w1L[i], 0.f) + 1e-3f;
  __syncthreads();

  // ---- WMMA GEMM: sim for all 5 ways, K split into two halves ----
  const int mt  = wv >> 1;              // M-tile 0..3 (two waves share one)
  const int ntp = (wv & 1) * 2;         // first of this wave's two N-tiles
  const int r   = lane & 15;
  const int hs  = lane >> 4;

  v8f acc[WAY][2];
#pragma unroll
  for (int w = 0; w < WAY; ++w)
#pragma unroll
    for (int t2 = 0; t2 < 2; ++t2)
#pragma unroll
      for (int j = 0; j < 8; ++j) acc[w][t2][j] = 0.f;

  _Float16* qnH = (_Float16*)smemRaw;
  const _Float16* pB0 = pnT + (size_t)tt * WAY * M * D;

  for (int kh = 0; kh < 2; ++kh) {
    __syncthreads();
    for (int dl = dg; dl < KHALF; dl += 4) {        // build normalized f16 tile
      int d = kh * KHALF + dl;
      qnH[m * QPITCH + dl] =
          (_Float16)(qf[qbase + (size_t)d * M + m] * rnormQ[m]);
    }
    __syncthreads();

    // A gather: lane hs=l>>4 holds contiguous K {16hs..16hs+15} (matches B)
    const _Float16* aRow = qnH + (16 * mt + r) * QPITCH + hs * 16;
#pragma unroll
    for (int w = 0; w < WAY; ++w) {
      const _Float16* bCol0 =
          pB0 + ((size_t)w * M + 16 * ntp + r) * D + kh * KHALF + hs * 16;
      const _Float16* bCol1 = bCol0 + (size_t)16 * D;
#pragma unroll
      for (int kk = 0; kk < 10; ++kk) {
        v8h alo = *(const v8h*)(aRow + kk * 32);
        v8h ahi = *(const v8h*)(aRow + kk * 32 + 8);
        v16h A = __builtin_shufflevector(alo, ahi, 0, 1, 2, 3, 4, 5, 6, 7, 8,
                                         9, 10, 11, 12, 13, 14, 15);
        v16h B0 = *(const v16h*)(bCol0 + kk * 32);
        v16h B1 = *(const v16h*)(bCol1 + kk * 32);
        acc[w][0] = __builtin_amdgcn_wmma_f32_16x16x32_f16(
            false, A, false, B0, (short)0, acc[w][0], false, false);
        acc[w][1] = __builtin_amdgcn_wmma_f32_16x16x32_f16(
            false, A, false, B1, (short)0, acc[w][1], false, false);
      }
    }
  }

  // ---- per way: dump sim/K, marginals, Sinkhorn, logits ----
  float* simL = (float*)smemRaw;                 // overlays qn tile
  float* KL   = simL + M * SPITCH;
  const float* pnormB = pnorm + (size_t)tt * WAY * M;

#pragma unroll
  for (int w = 0; w < WAY; ++w) {
    __syncthreads();
    // b-marginal partials: <pn[n,:], qmean>, 4 chunks of 160 over d
    {
      int nn = tid & 63, c = tid >> 6;
      const _Float16* prow = pB0 + ((size_t)w * M + nn) * D + c * 160;
      float p = 0.f;
      for (int i = 0; i < 160; i += 8) {
        v8h x = *(const v8h*)(prow + i);
#pragma unroll
        for (int j = 0; j < 8; ++j) p += (float)x[j] * qmeanL[c * 160 + i + j];
      }
      partL[c * 64 + nn] = p;
    }
    // dump this way's sim and K = exp(20*(sim-1)) (C-layout: M=8*hs+j, N=r)
#pragma unroll
    for (int t2 = 0; t2 < 2; ++t2) {
      int nt = ntp + t2;
#pragma unroll
      for (int j = 0; j < 8; ++j) {
        int mm = 16 * mt + 8 * hs + j;
        int nn = 16 * nt + r;
        float s = acc[w][t2][j];
        simL[mm * SPITCH + nn] = s;
        KL[mm * SPITCH + nn]   = exp2f((s - 1.0f) * 28.853901817f);
      }
    }
    if (tid < M) uL[tid] = 1.f;
    __syncthreads();

    if (tid == 0) {                        // finalize marginal sums (tiny)
      float sb = 0.f, sa = 0.f;
      for (int n2 = 0; n2 < M; ++n2) {
        float dot = (partL[n2] + partL[64 + n2] + partL[128 + n2] +
                     partL[192 + n2]) * (pnormB[w * M + n2] + 1e-8f);
        float w2 = fmaxf(dot, 0.f) + 1e-3f;
        bbuf[n2] = w2; sb += w2;
        sa += w1L[w * M + n2];
      }
      redL[0] = sb; redL[1] = sa;
    }
    __syncthreads();
    if (tid < M) {
      bbuf[tid] = bbuf[tid] / redL[0];
      abuf[tid] = w1L[w * M + tid] / redL[1];
    }
    __syncthreads();

    // Sinkhorn: u0=1; 100x (v = b/K^T u ; u = a/K v); final v
    for (int it = 0; it <= 100; ++it) {
      {
        int nn = tid & 63, c = tid >> 6;
        const float* Kc = KL + (16 * c) * SPITCH + nn;
        float p = 0.f;
#pragma unroll
        for (int i = 0; i < 16; ++i) p += Kc[i * SPITCH] * uL[16 * c + i];
        partL[c * 64 + nn] = p;
      }
      __syncthreads();
      if (tid < M) {
        float s = partL[tid] + partL[64 + tid] + partL[128 + tid] +
                  partL[192 + tid];
        vL[tid] = bbuf[tid] / (s + 1e-12f);
      }
      __syncthreads();
      if (it == 100) break;
      {
        int mm = tid & 63, c = tid >> 6;
        const float* Kr = KL + mm * SPITCH + 16 * c;
        float p = 0.f;
#pragma unroll
        for (int i = 0; i < 16; ++i) p += Kr[i] * vL[16 * c + i];
        partL[c * 64 + mm] = p;
      }
      __syncthreads();
      if (tid < M) {
        float s = partL[tid] + partL[64 + tid] + partL[128 + tid] +
                  partL[192 + tid];
        uL[tid] = abuf[tid] / (s + 1e-12f);
      }
      __syncthreads();
    }

    // logits = sum u_m K_mn v_n sim_mn
    {
      int mm = tid & 63, c = tid >> 6;
      float p = 0.f;
#pragma unroll
      for (int i = 0; i < 16; ++i) {
        int nn = 16 * c + i;
        p += KL[mm * SPITCH + nn] * simL[mm * SPITCH + nn] * vL[nn];
      }
      p *= uL[mm];
#pragma unroll
      for (int k = 16; k >= 1; k >>= 1) p += __shfl_xor(p, k, 32);
      if (lane == 0) redL[wv] = p;
    }
    __syncthreads();
    if (tid == 0) {
      float s = 0.f;
      for (int i = 0; i < 8; ++i) s += redL[i];
      out[(size_t)bid * WAY + w] = s;
    }
  }
}

extern "C" void kernel_launch(void* const* d_in, const int* in_sizes, int n_in,
                              void* d_out, int out_size, void* d_ws,
                              size_t ws_size, hipStream_t stream) {
  const float* qf = (const float*)d_in[0];
  const float* sf = (const float*)d_in[1];
  const int T = in_sizes[0] / (NQ * C);          // = 8 for the reference setup

  // workspace layout (≈3.39 MB): pnT f16 | pnorm f32 | pmean f32
  _Float16* pnT = (_Float16*)d_ws;
  float* pnorm  = (float*)((char*)d_ws + (size_t)T * WAY * M * D * sizeof(_Float16));
  float* pmean  = pnorm + (size_t)T * WAY * M;

  prep_support<<<T * WAY, 256, 0, stream>>>(sf, pnT, pnorm, pmean);
  emd_main<<<T * NQ, 256, 0, stream>>>(qf, pnT, pnorm, pmean, (float*)d_out);
}
